// GGL_26645977104434
// MI455X (gfx1250) — matrix operand order repaired and checked by reading, not compiled, not run on hardware
//
#include <hip/hip_runtime.h>
#include <hip/hip_bf16.h>

#define NSENS 4096
#define FEATD 2048
#define EPSV  1e-8f

#define BM 128
#define BN 128
#define BK 32

typedef __attribute__((ext_vector_type(16))) __bf16 v16bf;
typedef __attribute__((ext_vector_type(8)))  __bf16 v8bf;
typedef __attribute__((ext_vector_type(8)))  float  v8f;

// ---------------- CDNA5 async global->LDS copy (ASYNCcnt path) ----------------
// Per lane: LDS[lds_addr] = MEM[gaddr] for 16 bytes; tracked by ASYNCcnt.
__device__ __forceinline__ void async_copy_b128(unsigned lds_addr, const void* gptr) {
    asm volatile("global_load_async_to_lds_b128 %0, %1, off"
                 :: "v"(lds_addr), "v"((unsigned long long)(size_t)gptr)
                 : "memory");
}
__device__ __forceinline__ void wait_asynccnt0() {
    asm volatile("s_wait_asynccnt 0x0" ::: "memory");
}
// Flat shared-pointer -> 32-bit LDS byte address (low 32 bits of the flat address)
__device__ __forceinline__ unsigned lds_addr_of(const void* p) {
    return (unsigned)(size_t)p;
}

// ---------------- conversion: fp32 -> bf16 ----------------
__global__ void cvt_f32_bf16(const float* __restrict__ in, __bf16* __restrict__ out, int n) {
    int i = blockIdx.x * blockDim.x + threadIdx.x;
    if (i < n) out[i] = (__bf16)in[i];
}

// ---------------- W [K x N] -> W^T [N x K] as bf16 (LDS tiled) ----------------
__global__ void transpose_to_bf16(const float* __restrict__ W, __bf16* __restrict__ WT,
                                  int rows /*K*/, int cols /*N*/) {
    __shared__ float tile[32][33];
    int c0 = blockIdx.x * 32;
    int r0 = blockIdx.y * 32;
#pragma unroll
    for (int i = 0; i < 4; ++i)
        tile[threadIdx.y + i * 8][threadIdx.x] =
            W[(size_t)(r0 + threadIdx.y + i * 8) * cols + (c0 + threadIdx.x)];
    __syncthreads();
#pragma unroll
    for (int i = 0; i < 4; ++i)
        WT[(size_t)(c0 + threadIdx.y + i * 8) * rows + (r0 + threadIdx.x)] =
            (__bf16)tile[threadIdx.x][threadIdx.y + i * 8];
}

// ---------------- WMMA fragment loaders from LDS tiles (ld = BK) ----------------
// A 16x32 bf16: lane l(<16): M=row0+l, K=half*8+[0..7] and 16+half*8+[0..7]
__device__ __forceinline__ v16bf lds_frag_a(const __bf16* s, int row0, int lane) {
    int half = lane >> 4, l = lane & 15;
    const __bf16* p = s + (row0 + l) * BK + half * 8;
    v8bf lo = *(const v8bf*)p;
    v8bf hi = *(const v8bf*)(p + 16);
    v16bf a;
#pragma unroll
    for (int i = 0; i < 8; ++i) { a[i] = lo[i]; a[i + 8] = hi[i]; }
    return a;
}

// B 32x16 bf16: lane l(<16): N=col0+l, K=half*8+[0..7] and 16+half*8+[0..7]
__device__ __forceinline__ v16bf lds_frag_b(const __bf16* s, int col0, int lane) {
    int half = lane >> 4, l = lane & 15;
    const __bf16* p = s + (col0 + l) * BK + half * 8;
    v8bf lo = *(const v8bf*)p;
    v8bf hi = *(const v8bf*)(p + 16);
    v16bf b;
#pragma unroll
    for (int i = 0; i < 8; ++i) { b[i] = lo[i]; b[i + 8] = hi[i]; }
    return b;
}

// ---------------- LDS-tiled, register-blocked WMMA GEMM core ----------------
// C[tileM..+127, tileN..+127] += A[tileM..,:K] * BT[tileN..,:K]^T
// 8 waves: 2 (M) x 4 (N); each wave: 64x32 = 4x2 WMMA tiles.
// Tiles staged global->LDS with async copies (no VGPR staging), double-buffered.
__device__ __forceinline__ void gemm_core(const __bf16* __restrict__ A,
                                          const __bf16* __restrict__ BT,
                                          int K, int tileM, int tileN,
                                          v8f (&acc)[4][2]) {
    __shared__ __bf16 sA[2][BM * BK];   // 2 x 8 KB
    __shared__ __bf16 sB[2][BN * BK];   // 2 x 8 KB

    const int lane = threadIdx.x;
    const int wid  = threadIdx.y;
    const int t    = wid * 32 + lane;           // 0..255
    const int wr   = wid >> 2;                  // 0..1  (M)
    const int wc   = wid & 3;                   // 0..3  (N)
    const int wm0  = wr * 64;
    const int wn0  = wc * 32;

    // staging: 512 chunks of 16B per tile; 2 chunks per thread per matrix
    const int c0 = t, c1 = t + 256;
    const int ar0 = c0 >> 2, ak0 = (c0 & 3) * 8;   // 4 chunks per 32-elem row
    const int ar1 = c1 >> 2, ak1 = (c1 & 3) * 8;

    const __bf16* Ag = A  + (size_t)tileM * K;
    const __bf16* Bg = BT + (size_t)tileN * K;

    // async-load K-tile 0
    async_copy_b128(lds_addr_of(&sA[0][c0 * 8]), Ag + (size_t)ar0 * K + ak0);
    async_copy_b128(lds_addr_of(&sA[0][c1 * 8]), Ag + (size_t)ar1 * K + ak1);
    async_copy_b128(lds_addr_of(&sB[0][c0 * 8]), Bg + (size_t)ar0 * K + ak0);
    async_copy_b128(lds_addr_of(&sB[0][c1 * 8]), Bg + (size_t)ar1 * K + ak1);
    wait_asynccnt0();
    __syncthreads();

    const int KT = K / BK;
    for (int kt = 0; kt < KT; ++kt) {
        const int cur = kt & 1, nxt = cur ^ 1;

        // issue async copies of the next K-tile straight into the other LDS buffer;
        // they run on the async path while the WMMAs below consume sA/sB[cur]
        if (kt + 1 < KT) {
            const int k0 = (kt + 1) * BK;
            async_copy_b128(lds_addr_of(&sA[nxt][c0 * 8]), Ag + (size_t)ar0 * K + k0 + ak0);
            async_copy_b128(lds_addr_of(&sA[nxt][c1 * 8]), Ag + (size_t)ar1 * K + k0 + ak1);
            async_copy_b128(lds_addr_of(&sB[nxt][c0 * 8]), Bg + (size_t)ar0 * K + k0 + ak0);
            async_copy_b128(lds_addr_of(&sB[nxt][c1 * 8]), Bg + (size_t)ar1 * K + k0 + ak1);
        }
        // prefetch K-tile kt+2 toward L2 (global_prefetch_b8 path)
        if (kt + 2 < KT) {
            const int k2 = (kt + 2) * BK;
            __builtin_prefetch((const void*)(Ag + (size_t)ar0 * K + k2 + ak0), 0, 1);
            __builtin_prefetch((const void*)(Bg + (size_t)ar0 * K + k2 + ak0), 0, 1);
        }

        // fragments from LDS, then 8 WMMAs
        v16bf am[4], bn[2];
#pragma unroll
        for (int mi = 0; mi < 4; ++mi) am[mi] = lds_frag_a(sA[cur], wm0 + mi * 16, lane);
#pragma unroll
        for (int ni = 0; ni < 2; ++ni) bn[ni] = lds_frag_b(sB[cur], wn0 + ni * 16, lane);
#pragma unroll
        for (int mi = 0; mi < 4; ++mi)
#pragma unroll
            for (int ni = 0; ni < 2; ++ni)
                acc[mi][ni] = __builtin_amdgcn_wmma_f32_16x16x32_bf16(
                    false, am[mi], false, bn[ni], (short)0, acc[mi][ni], false, false);

        // this wave's async copies must have landed before everyone flips buffers
        if (kt + 1 < KT) wait_asynccnt0();
        __syncthreads();
    }
}

// ---------------- GEMM1: attr = sigmoid(x @ W + b), bf16 out ----------------
__global__ void __launch_bounds__(256)
gemm_sigmoid_bf16(const __bf16* __restrict__ A,   // [NSENS x FEATD]
                  const __bf16* __restrict__ BT,  // [NSENS x FEATD] = W^T
                  const float*  __restrict__ bias,
                  __bf16* __restrict__ out) {     // [NSENS x NSENS]
    v8f acc[4][2];
#pragma unroll
    for (int mi = 0; mi < 4; ++mi)
#pragma unroll
        for (int ni = 0; ni < 2; ++ni)
#pragma unroll
            for (int r = 0; r < 8; ++r) acc[mi][ni][r] = 0.f;

    const int tileM = blockIdx.y * BM, tileN = blockIdx.x * BN;
    gemm_core(A, BT, FEATD, tileM, tileN, acc);

    const int lane = threadIdx.x, wid = threadIdx.y;
    const int wr = wid >> 2, wc = wid & 3;
    const int half = lane >> 4, l = lane & 15;
#pragma unroll
    for (int ni = 0; ni < 2; ++ni) {
        const int col = tileN + wc * 32 + ni * 16 + l;
        const float bv = bias[col];
#pragma unroll
        for (int mi = 0; mi < 4; ++mi) {
#pragma unroll
            for (int r = 0; r < 8; ++r) {
                const int row = tileM + wr * 64 + mi * 16 + r + half * 8;
                const float v = acc[mi][ni][r] + bv;
                out[(size_t)row * NSENS + col] = (__bf16)(1.0f / (1.0f + __expf(-v)));
            }
        }
    }
}

// ---------------- row L2 norms of attr ----------------
__global__ void row_norms(const __bf16* __restrict__ attr, float* __restrict__ norms) {
    __shared__ float red[256];
    int row = blockIdx.x, tid = threadIdx.x;
    float s = 0.f;
    for (int i = tid; i < NSENS; i += 256) {
        float v = (float)attr[(size_t)row * NSENS + i];
        s += v * v;
    }
    red[tid] = s;
    __syncthreads();
    for (int off = 128; off > 0; off >>= 1) {
        if (tid < off) red[tid] += red[tid + off];
        __syncthreads();
    }
    if (tid == 0) norms[row] = fmaxf(sqrtf(red[0]), EPSV);
}

// ---------------- GEMM2: adj = (attr @ attr^T) / (n_i * n_j), fp32 out ----------------
__global__ void __launch_bounds__(256)
gemm_cosine(const __bf16* __restrict__ attr, const float* __restrict__ norms,
            float* __restrict__ adj) {
    v8f acc[4][2];
#pragma unroll
    for (int mi = 0; mi < 4; ++mi)
#pragma unroll
        for (int ni = 0; ni < 2; ++ni)
#pragma unroll
            for (int r = 0; r < 8; ++r) acc[mi][ni][r] = 0.f;

    const int tileM = blockIdx.y * BM, tileN = blockIdx.x * BN;
    gemm_core(attr, attr, NSENS, tileM, tileN, acc);   // B col c = attr row c (attr^T)

    const int lane = threadIdx.x, wid = threadIdx.y;
    const int wr = wid >> 2, wc = wid & 3;
    const int half = lane >> 4, l = lane & 15;
#pragma unroll
    for (int ni = 0; ni < 2; ++ni) {
        const int col = tileN + wc * 32 + ni * 16 + l;
        const float ninv = 1.0f / norms[col];
#pragma unroll
        for (int mi = 0; mi < 4; ++mi) {
#pragma unroll
            for (int r = 0; r < 8; ++r) {
                const int row = tileM + wr * 64 + mi * 16 + r + half * 8;
                adj[(size_t)row * NSENS + col] = acc[mi][ni][r] * ninv / norms[row];
            }
        }
    }
}

// ---------------- edge_index src half: row ids ----------------
__global__ void fill_src(int* __restrict__ src, int total) {
    int i = blockIdx.x * blockDim.x + threadIdx.x;
    if (i < total) src[i] = i >> 12; // i / NSENS
}

// ---------------- per-row: rowmax -> normalize (A_norm) -> bitonic sort desc ----------------
__global__ void rownorm_sort(float* __restrict__ adj,     // in: cosine; out: A_norm (in place)
                             float* __restrict__ values,
                             int*   __restrict__ indices) {
    __shared__ float sv[NSENS];
    __shared__ int   si[NSENS];
    __shared__ float red[256];
    int row = blockIdx.x, tid = threadIdx.x;
    float* rowp = adj + (size_t)row * NSENS;

    float m = -1e30f;
    for (int i = tid; i < NSENS; i += 256) {
        float v = rowp[i];
        sv[i] = v;
        m = fmaxf(m, v);
    }
    red[tid] = m;
    __syncthreads();
    for (int off = 128; off > 0; off >>= 1) {
        if (tid < off) red[tid] = fmaxf(red[tid], red[tid + off]);
        __syncthreads();
    }
    float rinv = 1.0f / red[0];

    for (int i = tid; i < NSENS; i += 256) {
        float v = sv[i] * rinv;
        sv[i] = v;
        si[i] = i;
        rowp[i] = v;                       // A_norm output
    }

    for (int k = 2; k <= NSENS; k <<= 1) {
        for (int j = k >> 1; j > 0; j >>= 1) {
            __syncthreads();
            for (int t = tid; t < NSENS / 2; t += 256) {
                int i   = ((t & ~(j - 1)) << 1) | (t & (j - 1));
                int ixj = i | j;
                bool desc = ((i & k) == 0);
                float a = sv[i], b = sv[ixj];
                if ((a < b) == desc) {
                    sv[i] = b; sv[ixj] = a;
                    int ti = si[i]; si[i] = si[ixj]; si[ixj] = ti;
                }
            }
        }
    }
    __syncthreads();
    for (int i = tid; i < NSENS; i += 256) {
        values[(size_t)row * NSENS + i]  = sv[i];
        indices[(size_t)row * NSENS + i] = si[i];
    }
}

extern "C" void kernel_launch(void* const* d_in, const int* in_sizes, int n_in,
                              void* d_out, int out_size, void* d_ws, size_t ws_size,
                              hipStream_t stream) {
    const float* x = (const float*)d_in[0];   // [NSENS, FEATD]
    const float* W = (const float*)d_in[1];   // [FEATD, NSENS]
    const float* b = (const float*)d_in[2];   // [NSENS]

    const size_t NN = (size_t)NSENS * NSENS;
    float* out    = (float*)d_out;
    float* values = out;                      // n*n
    int*   srcIdx = (int*)(out + NN);         // n*n (edge_index row 0)
    int*   dstIdx = (int*)(out + 2 * NN);     // n*n (edge_index row 1)
    float* Anorm  = out + 3 * NN;             // n*n

    char* ws = (char*)d_ws;
    __bf16* xh    = (__bf16*)(ws);                          // 16 MB
    __bf16* WT    = (__bf16*)(ws + ((size_t)16 << 20));     // 16 MB
    __bf16* attr  = (__bf16*)(ws + ((size_t)32 << 20));     // 32 MB
    float*  norms = (float*)(ws + ((size_t)64 << 20));      // 16 KB

    // 1) convert x to bf16
    {
        int n = NSENS * FEATD;
        cvt_f32_bf16<<<(n + 255) / 256, 256, 0, stream>>>(x, xh, n);
    }
    // 2) W [FEATD x NSENS] -> W^T [NSENS x FEATD] bf16
    {
        dim3 blk(32, 8), grd(NSENS / 32, FEATD / 32);
        transpose_to_bf16<<<grd, blk, 0, stream>>>(W, WT, FEATD, NSENS);
    }
    // 3) attr = sigmoid(x @ W + b)  (LDS-tiled WMMA, async staging)
    {
        dim3 blk(32, 8), grd(NSENS / BN, NSENS / BM);       // (32, 32)
        gemm_sigmoid_bf16<<<grd, blk, 0, stream>>>(xh, WT, b, attr);
    }
    // 4) row norms
    row_norms<<<NSENS, 256, 0, stream>>>(attr, norms);
    // 5) adj = cosine-similarity gram matrix -> Anorm region (LDS-tiled WMMA)
    {
        dim3 blk(32, 8), grd(NSENS / BN, NSENS / BM);
        gemm_cosine<<<grd, blk, 0, stream>>>(attr, norms, Anorm);
    }
    // 6) edge_index src half
    fill_src<<<(int)(NN / 256), 256, 0, stream>>>(srcIdx, (int)NN);
    // 7) row-max normalize (in place -> A_norm) + full-row descending sort
    rownorm_sort<<<NSENS, 256, 0, stream>>>(Anorm, values, dstIdx);
}
